// RecurrentAE_3736621547811
// MI455X (gfx1250) — compile-verified, matrix operand order checked by reading, not run on hardware
//
#include <hip/hip_runtime.h>
#include <hip/hip_bf16.h>

// Problem constants (match reference): B=512, S=1024, F=64, L=128
#define BATCH 512
#define SEQ   1024
#define NF    64
#define NL    128
#define NG    512   // 4*NL gate columns
#define KDIM  192   // NF + NL concat-K

typedef __bf16 bf16_t;
typedef bf16_t v16bf __attribute__((ext_vector_type(16)));
typedef float  v8f   __attribute__((ext_vector_type(8)));

static __device__ __forceinline__ v8f wmma_bf16(v16bf a, v16bf b, v8f c) {
  // D = A(16x32 bf16) x B(32x16 bf16) + C(16x16 f32)
  return __builtin_amdgcn_wmma_f32_16x16x32_bf16(false, a, false, b, (short)0, c, false, false);
}

static __device__ __forceinline__ v8f bcast8(float v) {
  v8f r;
  #pragma unroll
  for (int i = 0; i < 8; ++i) r[i] = v;
  return r;
}

// A/B fragment per lane: 8 contiguous bf16 at koff, 8 more at koff+16
// (lane<16 -> koff = kbase; lane>=16 -> koff = kbase+8), per 16-bit 16x32 layout.
static __device__ __forceinline__ v16bf frag_from_lds(const bf16_t* rowp, int koff) {
  union { v16bf v; uint4 u[2]; } t;
  t.u[0] = *(const uint4*)(rowp + koff);
  t.u[1] = *(const uint4*)(rowp + koff + 16);
  return t.v;
}

static __device__ __forceinline__ v16bf frag_from_global_f32(const float* p, int koff) {
  const float4 a0 = *(const float4*)(p + koff);
  const float4 a1 = *(const float4*)(p + koff + 4);
  const float4 b0 = *(const float4*)(p + koff + 16);
  const float4 b1 = *(const float4*)(p + koff + 20);
  v16bf r;
  r[0]=(bf16_t)a0.x;  r[1]=(bf16_t)a0.y;  r[2]=(bf16_t)a0.z;  r[3]=(bf16_t)a0.w;
  r[4]=(bf16_t)a1.x;  r[5]=(bf16_t)a1.y;  r[6]=(bf16_t)a1.z;  r[7]=(bf16_t)a1.w;
  r[8]=(bf16_t)b0.x;  r[9]=(bf16_t)b0.y;  r[10]=(bf16_t)b0.z; r[11]=(bf16_t)b0.w;
  r[12]=(bf16_t)b1.x; r[13]=(bf16_t)b1.y; r[14]=(bf16_t)b1.z; r[15]=(bf16_t)b1.w;
  return r;
}

// gfx1250 has a hardware v_tanh_f32 (single TRANS op). Fall back to exp-based
// tanh if the builtin is unavailable on this toolchain.
#if __has_builtin(__builtin_amdgcn_tanhf)
static __device__ __forceinline__ float fast_tanh(float x) {
  return __builtin_amdgcn_tanhf(x);
}
#else
static __device__ __forceinline__ float fast_tanh(float x) {
  const float e = __expf(-2.0f * __builtin_fabsf(x));
  const float t = (1.0f - e) * __builtin_amdgcn_rcpf(1.0f + e);
  return __builtin_copysignf(t, x);
}
#endif
// sigmoid(x) = 0.5*tanh(0.5x) + 0.5  (one TRANS + one FMA)
static __device__ __forceinline__ float sigm(float x) {
  return __builtin_fmaf(0.5f, fast_tanh(0.5f * x), 0.5f);
}

// Pack 4 floats -> 4 bf16 and store as one 64-bit LDS write (addr 8B-aligned).
static __device__ __forceinline__ void store4_bf16(bf16_t* p, float a, float b, float c, float d) {
  union { uint2 u; bf16_t h[4]; } t;
  t.h[0] = (bf16_t)a; t.h[1] = (bf16_t)b; t.h[2] = (bf16_t)c; t.h[3] = (bf16_t)d;
  *(uint2*)p = t.u;
}
static __device__ __forceinline__ void store2_bf16(bf16_t* p, float a, float b) {
  union { unsigned u; bf16_t h[2]; } t;
  t.h[0] = (bf16_t)a; t.h[1] = (bf16_t)b;
  *(unsigned*)p = t.u;
}

extern "C" __global__ __launch_bounds__(512, 1)
void rae_kernel(const float* __restrict__ x,
                const float* __restrict__ WihE, const float* __restrict__ WhhE,
                const float* __restrict__ bihE, const float* __restrict__ bhhE,
                const float* __restrict__ WihD, const float* __restrict__ WhhD,
                const float* __restrict__ bihD, const float* __restrict__ bhhD,
                const float* __restrict__ Wdn,  const float* __restrict__ bdn,
                float* __restrict__ out)
{
  // Dynamic per-tile state only; weights + biases + cell state live in registers.
  __shared__ __align__(16) bf16_t ah[16 * KDIM];   // [x_t(0..63) ; h(64..191)] bf16
  __shared__ float gbuf[16 * NG];                  // gate pre-activations

  const int tid  = threadIdx.x;
  const int lane = tid & 31;
  const int w    = tid >> 5;          // wave id 0..15
  const int l15  = lane & 15;
  const int sel  = (lane >> 4) << 3;  // K sub-select per 16x32 bf16 layout
  const int mofs = (lane >> 4) << 3;  // D-tile row offset (lanes 16-31 -> M+8)
  const int rowbase = blockIdx.x << 4;

  const int m  = tid >> 5;            // batch row handled in elementwise phase
  const int j0 = (tid & 31) << 2;     // 4 latent columns per thread

  const int col0 = (w << 5) + l15;    // first gate column owned by this lane
  const int nd   = ((w & 3) << 4) + l15;

  // ---- per-lane bias registers (folded into WMMA accumulator init) ----
  const float bE0 = bihE[col0]      + bhhE[col0];
  const float bE1 = bihE[col0 + 16] + bhhE[col0 + 16];
  const float bD0 = bihD[col0]      + bhhD[col0];
  const float bD1 = bihD[col0 + 16] + bhhD[col0 + 16];
  const float bv  = bdn[nd];

  // ---- cell state in registers (thread->(m,j) mapping is step-invariant) ----
  float creg[4] = {0.0f, 0.0f, 0.0f, 0.0f};

  // ---- init: zero h, stage x_0 ----
  store4_bf16(&ah[m * KDIM + NF + j0], 0.0f, 0.0f, 0.0f, 0.0f);
  {
    const int f2 = (tid & 31) << 1;
    const float2 v = *(const float2*)(x + ((size_t)(rowbase + m) * SEQ) * NF + f2);
    store2_bf16(&ah[m * KDIM + f2], v.x, v.y);
  }

  // ---- encoder weight fragments into registers (fp32 -> bf16) ----
  v16bf Wb[2][6];
  #pragma unroll
  for (int tile = 0; tile < 2; ++tile) {
    const int n = (w << 5) + (tile << 4) + l15;
    const float* rih = WihE + n * NF;
    const float* rhh = WhhE + n * NL;
    #pragma unroll
    for (int kc = 0; kc < 2; ++kc) Wb[tile][kc]     = frag_from_global_f32(rih, kc * 32 + sel);
    #pragma unroll
    for (int kc = 0; kc < 4; ++kc) Wb[tile][2 + kc] = frag_from_global_f32(rhh, kc * 32 + sel);
  }
  // dense (decoder output) weight fragments
  v16bf Wdf[4];
  {
    const float* rdn = Wdn + nd * NL;
    #pragma unroll
    for (int kc = 0; kc < 4; ++kc) Wdf[kc] = frag_from_global_f32(rdn, kc * 32 + sel);
  }

  const bf16_t* arow = ah + l15 * KDIM;

  __syncthreads();

  // ================= encoder: 1024 sequential steps =================
  #pragma unroll 1
  for (int t = 0; t < SEQ; ++t) {
    // Issue ALL A-fragment loads first (distinct regs -> pipelined ds_loads),
    // then drain into the WMMA chain.
    v16bf afr[6];
    #pragma unroll
    for (int kc = 0; kc < 6; ++kc) afr[kc] = frag_from_lds(arow, kc * 32 + sel);

    v8f acc0 = bcast8(bE0);
    v8f acc1 = bcast8(bE1);
    #pragma unroll
    for (int kc = 0; kc < 6; ++kc) {
      acc0 = wmma_bf16(afr[kc], Wb[0][kc], acc0);
      acc1 = wmma_bf16(afr[kc], Wb[1][kc], acc1);
    }
    #pragma unroll
    for (int r = 0; r < 8; ++r) {
      gbuf[(mofs + r) * NG + col0]      = acc0[r];
      gbuf[(mofs + r) * NG + col0 + 16] = acc1[r];
    }
    __syncthreads();

    // LSTM cell elementwise, staged for ILP across the 4 columns.
    float ig[4], fg[4], gg[4], og[4];
    #pragma unroll
    for (int q = 0; q < 4; ++q) {
      const int j = j0 + q;
      ig[q] = gbuf[m * NG + j];
      fg[q] = gbuf[m * NG + j + NL];
      gg[q] = gbuf[m * NG + j + 2 * NL];
      og[q] = gbuf[m * NG + j + 3 * NL];
    }
    float si[4], sf[4], so[4], tg[4];
    #pragma unroll
    for (int q = 0; q < 4; ++q) si[q] = sigm(ig[q]);
    #pragma unroll
    for (int q = 0; q < 4; ++q) sf[q] = sigm(fg[q]);
    #pragma unroll
    for (int q = 0; q < 4; ++q) so[q] = sigm(og[q]);
    #pragma unroll
    for (int q = 0; q < 4; ++q) tg[q] = fast_tanh(gg[q]);
    float hv[4];
    #pragma unroll
    for (int q = 0; q < 4; ++q) {
      const float c = sf[q] * creg[q] + si[q] * tg[q];
      creg[q] = c;
      hv[q] = so[q] * fast_tanh(c);
    }
    store4_bf16(&ah[m * KDIM + NF + j0], hv[0], hv[1], hv[2], hv[3]);

    // stage x_{t+1}, prefetch x_{t+2}
    if (t + 1 < SEQ) {
      const int f2 = (tid & 31) << 1;
      const float* px = x + ((size_t)(rowbase + m) * SEQ + (t + 1)) * NF + f2;
      const float2 v = *(const float2*)px;
      store2_bf16(&ah[m * KDIM + f2], v.x, v.y);
      if (t + 2 < SEQ) __builtin_prefetch(px + NF, 0, 1);
    }
    __syncthreads();
  }

  // ---- swap to decoder weight fragments ----
  #pragma unroll
  for (int tile = 0; tile < 2; ++tile) {
    const int n = (w << 5) + (tile << 4) + l15;
    const float* rih = WihD + n * NF;
    const float* rhh = WhhD + n * NL;
    #pragma unroll
    for (int kc = 0; kc < 2; ++kc) Wb[tile][kc]     = frag_from_global_f32(rih, kc * 32 + sel);
    #pragma unroll
    for (int kc = 0; kc < 4; ++kc) Wb[tile][2 + kc] = frag_from_global_f32(rhh, kc * 32 + sel);
  }

  // x0 = h_n @ Wd^T + b_dense (feeds decoder, not written to output)
  if (w < 4) {
    v16bf afr[4];
    #pragma unroll
    for (int kc = 0; kc < 4; ++kc) afr[kc] = frag_from_lds(arow + NF, kc * 32 + sel);
    v8f acc = bcast8(bv);
    #pragma unroll
    for (int kc = 0; kc < 4; ++kc) acc = wmma_bf16(afr[kc], Wdf[kc], acc);
    #pragma unroll
    for (int r = 0; r < 8; ++r)
      ah[(mofs + r) * KDIM + nd] = (bf16_t)acc[r];
  }
  __syncthreads();

  // ================= decoder: 1024 autoregressive steps =================
  #pragma unroll 1
  for (int t = 0; t < SEQ; ++t) {
    v16bf afr[6];
    #pragma unroll
    for (int kc = 0; kc < 6; ++kc) afr[kc] = frag_from_lds(arow, kc * 32 + sel);

    v8f acc0 = bcast8(bD0);
    v8f acc1 = bcast8(bD1);
    #pragma unroll
    for (int kc = 0; kc < 6; ++kc) {
      acc0 = wmma_bf16(afr[kc], Wb[0][kc], acc0);
      acc1 = wmma_bf16(afr[kc], Wb[1][kc], acc1);
    }
    #pragma unroll
    for (int r = 0; r < 8; ++r) {
      gbuf[(mofs + r) * NG + col0]      = acc0[r];
      gbuf[(mofs + r) * NG + col0 + 16] = acc1[r];
    }
    __syncthreads();

    float ig[4], fg[4], gg[4], og[4];
    #pragma unroll
    for (int q = 0; q < 4; ++q) {
      const int j = j0 + q;
      ig[q] = gbuf[m * NG + j];
      fg[q] = gbuf[m * NG + j + NL];
      gg[q] = gbuf[m * NG + j + 2 * NL];
      og[q] = gbuf[m * NG + j + 3 * NL];
    }
    float si[4], sf[4], so[4], tg[4];
    #pragma unroll
    for (int q = 0; q < 4; ++q) si[q] = sigm(ig[q]);
    #pragma unroll
    for (int q = 0; q < 4; ++q) sf[q] = sigm(fg[q]);
    #pragma unroll
    for (int q = 0; q < 4; ++q) so[q] = sigm(og[q]);
    #pragma unroll
    for (int q = 0; q < 4; ++q) tg[q] = fast_tanh(gg[q]);
    float hv[4];
    #pragma unroll
    for (int q = 0; q < 4; ++q) {
      const float c = sf[q] * creg[q] + si[q] * tg[q];
      creg[q] = c;
      hv[q] = so[q] * fast_tanh(c);
    }
    store4_bf16(&ah[m * KDIM + NF + j0], hv[0], hv[1], hv[2], hv[3]);
    __syncthreads();

    // x_i = h @ Wd^T + b_dense; write flipped output, feed back as next input
    if (w < 4) {
      v16bf afr2[4];
      #pragma unroll
      for (int kc = 0; kc < 4; ++kc) afr2[kc] = frag_from_lds(arow + NF, kc * 32 + sel);
      v8f acc = bcast8(bv);
      #pragma unroll
      for (int kc = 0; kc < 4; ++kc) acc = wmma_bf16(afr2[kc], Wdf[kc], acc);
      const int sOut = SEQ - 1 - t;
      #pragma unroll
      for (int r = 0; r < 8; ++r) {
        const float v = acc[r];
        ah[(mofs + r) * KDIM + nd] = (bf16_t)v;
        out[((size_t)(rowbase + mofs + r) * SEQ + sOut) * NF + nd] = v;
      }
    }
    __syncthreads();
  }
}

extern "C" void kernel_launch(void* const* d_in, const int* in_sizes, int n_in,
                              void* d_out, int out_size, void* d_ws, size_t ws_size,
                              hipStream_t stream) {
  (void)in_sizes; (void)n_in; (void)d_ws; (void)ws_size; (void)out_size;
  const float* x    = (const float*)d_in[0];
  const float* WihE = (const float*)d_in[1];
  const float* WhhE = (const float*)d_in[2];
  const float* bihE = (const float*)d_in[3];
  const float* bhhE = (const float*)d_in[4];
  const float* WihD = (const float*)d_in[5];
  const float* WhhD = (const float*)d_in[6];
  const float* bihD = (const float*)d_in[7];
  const float* bhhD = (const float*)d_in[8];
  const float* Wdn  = (const float*)d_in[9];
  const float* bdn  = (const float*)d_in[10];
  float* out = (float*)d_out;

  dim3 grid(BATCH / 16);
  dim3 block(512);
  hipLaunchKernelGGL(rae_kernel, grid, block, 0, stream,
                     x, WihE, WhhE, bihE, bhhE, WihD, WhhD, bihD, bhhD, Wdn, bdn, out);
}
